// PatchedMBartSqueezeAttention_3298534883706
// MI455X (gfx1250) — matrix operand order repaired and checked by reading, not compiled, not run on hardware
//
#include <hip/hip_runtime.h>
#include <hip/hip_bf16.h>

typedef __attribute__((ext_vector_type(16))) _Float16 v16h;
typedef __attribute__((ext_vector_type(8)))  float    v8f;

// ---------------------------------------------------------------------------
// WMMA GEMM: Out[32 x N] = A[32 x K] @ W[N x K]^T + bias, then * scale.
// One wave per 16x16 output tile, K in chunks of 32 with
// v_wmma_f32_16x16x32_f16 (f32 accumulate). M fixed at 32 (2 M-tiles).
//
// Per-lane register layouts (CDNA5 ISA 7.12.2, wave32):
//   A 16x32 f16 : lane l -> row m=l&15, half h=l>>4;
//                 elems 0..7  = A[m][k0+8h .. +7], elems 8..15 = A[m][k0+16+8h .. +7]
//   B 32x16 f16 : lane l -> col n=l&15, half h=l>>4;
//                 elems 0..7  = B[k0+8h .. +7][n], elems 8..15 = B[k0+16+8h .. +7][n]  (== W[n][k...])
//   C/D 16x16 f32: VGPR r, lane l -> row r+8h, col l&15
// ---------------------------------------------------------------------------
__global__ __launch_bounds__(32) void gemm32_wmma_f16(
    const float* __restrict__ A,    // [32 x K] row-major
    const float* __restrict__ W,    // [N x K] row-major (acts as B^T)
    const float* __restrict__ bias, // [N]
    float* __restrict__ Out,        // [32 x N]
    int N, int K, float scale)
{
    int tile = blockIdx.x;
    int mt = (tile & 1) * 16;        // 2 M-tiles (M == 32)
    int nt = (tile >> 1) * 16;
    int lane = threadIdx.x & 31;
    int mr = lane & 15;
    int hh = lane >> 4;              // 0 or 1

    const float* arow = A + (size_t)(mt + mr) * K;
    const float* wrow = W + (size_t)(nt + mr) * K;

    v8f c = {};
#pragma unroll 2
    for (int k0 = 0; k0 < K; k0 += 32) {
        v16h av, bv;
        const float4* ap0 = (const float4*)(arow + k0 + 8 * hh);
        const float4* ap1 = (const float4*)(arow + k0 + 16 + 8 * hh);
        float4 a0 = ap0[0], a1 = ap0[1];
        float4 a2 = ap1[0], a3 = ap1[1];
        av[0]=(_Float16)a0.x; av[1]=(_Float16)a0.y; av[2]=(_Float16)a0.z; av[3]=(_Float16)a0.w;
        av[4]=(_Float16)a1.x; av[5]=(_Float16)a1.y; av[6]=(_Float16)a1.z; av[7]=(_Float16)a1.w;
        av[8]=(_Float16)a2.x; av[9]=(_Float16)a2.y; av[10]=(_Float16)a2.z; av[11]=(_Float16)a2.w;
        av[12]=(_Float16)a3.x; av[13]=(_Float16)a3.y; av[14]=(_Float16)a3.z; av[15]=(_Float16)a3.w;

        const float4* bp0 = (const float4*)(wrow + k0 + 8 * hh);
        const float4* bp1 = (const float4*)(wrow + k0 + 16 + 8 * hh);
        float4 b0 = bp0[0], b1 = bp0[1], b2 = bp1[0], b3 = bp1[1];
        bv[0]=(_Float16)b0.x; bv[1]=(_Float16)b0.y; bv[2]=(_Float16)b0.z; bv[3]=(_Float16)b0.w;
        bv[4]=(_Float16)b1.x; bv[5]=(_Float16)b1.y; bv[6]=(_Float16)b1.z; bv[7]=(_Float16)b1.w;
        bv[8]=(_Float16)b2.x; bv[9]=(_Float16)b2.y; bv[10]=(_Float16)b2.z; bv[11]=(_Float16)b2.w;
        bv[12]=(_Float16)b3.x; bv[13]=(_Float16)b3.y; bv[14]=(_Float16)b3.z; bv[15]=(_Float16)b3.w;

        c = __builtin_amdgcn_wmma_f32_16x16x32_f16(
                /*neg_a=*/false, av, /*neg_b=*/false, bv,
                /*c_mod=*/(short)0, c, /*reuse_a=*/false, /*reuse_b=*/false);
    }

#pragma unroll
    for (int r = 0; r < 8; ++r) {
        int row = mt + r + 8 * hh;
        int col = nt + mr;
        Out[(size_t)row * N + col] = (c[r] + bias[col]) * scale;
    }
}

// ---------------------------------------------------------------------------
// Fused flash-decode attention, one workgroup (8 wave32s) per (b,h).
// Streams past_key / past_value exactly once with b128-per-lane loads
// (512B contiguous per wave per instruction), scores staged in LDS,
// block softmax, with linear fix-up for the kvlen scatter row.
// ---------------------------------------------------------------------------
__global__ __launch_bounds__(256) void attn_decode(
    const float* __restrict__ qsc,   // [32 x 512]  scaled Q, col = h*32+d
    const float* __restrict__ knew,  // [32 x 512]
    const float* __restrict__ vnew,  // [32 x 1024] col = h*64+d
    const float* __restrict__ Kc,    // past_key   [32,16,4096,32]
    const float* __restrict__ Vc,    // past_value [32,16,4096,64]
    const float* __restrict__ mask,  // [32 x 4096]
    const int*   __restrict__ kvlen, // [1]
    float* __restrict__ attn)        // [32 x 1024] col = h*64+d
{
    constexpr int SEQ = 4096;
    __shared__ float sscore[SEQ];
    __shared__ float sq[32];
    __shared__ float red[256];
    __shared__ float sacc[16][64];
    __shared__ float spsum[16];

    int bh = blockIdx.x;
    int b = bh >> 4, h = bh & 15;
    int tid = threadIdx.x;
    int w = tid >> 5, lane = tid & 31;
    int kv = kvlen[0];

    if (tid < 32) sq[tid] = qsc[b * 512 + h * 32 + tid];
    __syncthreads();

    const float* Kbase = Kc + (size_t)bh * SEQ * 32;
    const float* Vbase = Vc + (size_t)bh * SEQ * 64;
    const float* mrow  = mask + (size_t)b * SEQ;

    // ---- phase 1: scores[s] = q . K[s] + mask[s] ----
    // 8 lanes x float4 cover one 32-float K row -> 4 rows / wave / iteration,
    // 512B contiguous per global_load_b128 wave-instruction.
    {
        int g = lane >> 3;            // row subgroup 0..3
        int e = lane & 7;             // float4 slot within row
        float q0 = sq[4 * e + 0], q1 = sq[4 * e + 1];
        float q2 = sq[4 * e + 2], q3 = sq[4 * e + 3];
        for (int s0 = 4 * w; s0 < SEQ; s0 += 32) {
            int row = s0 + g;
            float4 k4 = *(const float4*)(Kbase + (size_t)row * 32 + 4 * e);
            float d = q0 * k4.x + q1 * k4.y + q2 * k4.z + q3 * k4.w;
#pragma unroll
            for (int off = 4; off; off >>= 1) d += __shfl_xor(d, off, 32);
            if (e == 0) sscore[row] = d + mrow[row];
        }
    }
    __syncthreads();

    // scatter fix-up: row kv uses k_new instead of past_key
    if (tid < 32) {
        float d = sq[lane] * knew[b * 512 + h * 32 + lane];
#pragma unroll
        for (int off = 16; off; off >>= 1) d += __shfl_xor(d, off, 32);
        if (lane == 0) sscore[kv] = d + mrow[kv];
    }
    __syncthreads();

    // ---- row max reduction ----
    float m = -1e30f;
    for (int s = tid; s < SEQ; s += 256) m = fmaxf(m, sscore[s]);
    red[tid] = m;
    __syncthreads();
    for (int st = 128; st; st >>= 1) {
        if (tid < st) red[tid] = fmaxf(red[tid], red[tid + st]);
        __syncthreads();
    }
    float rmax = red[0];

    // ---- phase 2: acc[d] = sum_s exp(score-max) * V[s][d] ----
    // 16 lanes x float4 cover one 64-float V row -> 2 rows / wave / iteration.
    int g2 = lane >> 4;               // 0..1 row subgroup
    int e2 = lane & 15;               // float4 slot (col = 4*e2)
    float ax = 0.f, ay = 0.f, az = 0.f, aw = 0.f, psum = 0.f;
    for (int s0 = 2 * w; s0 < SEQ; s0 += 16) {
        int row = s0 + g2;
        float p = __expf(sscore[row] - rmax);
        psum += p;
        float4 v4 = *(const float4*)(Vbase + (size_t)row * 64 + 4 * e2);
        ax += p * v4.x; ay += p * v4.y; az += p * v4.z; aw += p * v4.w;
    }
    // scatter fix-up on V (linear in accumulation; one 16-lane group applies it)
    if (w == 0 && g2 == 0) {
        float pkv = __expf(sscore[kv] - rmax);
        float4 vo = *(const float4*)(Vbase + (size_t)kv * 64 + 4 * e2);
        float4 nv = *(const float4*)(vnew + b * 1024 + h * 64 + 4 * e2);
        ax += pkv * (nv.x - vo.x); ay += pkv * (nv.y - vo.y);
        az += pkv * (nv.z - vo.z); aw += pkv * (nv.w - vo.w);
    }
    int slot = w * 2 + g2;            // 16 partial accumulators
    sacc[slot][4 * e2 + 0] = ax;
    sacc[slot][4 * e2 + 1] = ay;
    sacc[slot][4 * e2 + 2] = az;
    sacc[slot][4 * e2 + 3] = aw;
    if (e2 == 0) spsum[slot] = psum;
    __syncthreads();

    if (tid < 64) {
        float a = 0.f, ts = 0.f;
#pragma unroll
        for (int i = 0; i < 16; ++i) { a += sacc[i][tid]; ts += spsum[i]; }
        attn[b * 1024 + h * 64 + tid] = a / ts;
    }
}

// ---------------------------------------------------------------------------
extern "C" void kernel_launch(void* const* d_in, const int* in_sizes, int n_in,
                              void* d_out, int out_size, void* d_ws, size_t ws_size,
                              hipStream_t stream) {
    const float* hidden     = (const float*)d_in[0];   // [32,1,1024]
    const float* past_key   = (const float*)d_in[1];   // [32,16,4096,32]
    const float* past_value = (const float*)d_in[2];   // [32,16,4096,64]
    const float* mask       = (const float*)d_in[3];   // [32,1,1,4096]
    const int*   kvlen      = (const int*)d_in[4];     // [1]
    const float* q_w = (const float*)d_in[5];
    const float* q_b = (const float*)d_in[6];
    const float* k_w = (const float*)d_in[7];
    const float* k_b = (const float*)d_in[8];
    const float* v_w = (const float*)d_in[9];
    const float* v_b = (const float*)d_in[10];
    const float* o_w = (const float*)d_in[11];
    const float* o_b = (const float*)d_in[12];
    float* out = (float*)d_out;

    float* ws   = (float*)d_ws;
    float* qs   = ws;          // 32*512  = 16384 floats
    float* kn   = ws + 16384;  // 32*512
    float* vn   = ws + 32768;  // 32*1024 = 32768 floats
    float* attn = ws + 65536;  // 32*1024   (total 384 KB)

    const float SCALING = 0.17677669529663687f;  // 32^-0.5

    // Projections: M=32, K=1024; one wave per 16x16 tile.
    gemm32_wmma_f16<<<2 * (512  / 16), 32, 0, stream>>>(hidden, q_w, q_b, qs, 512,  1024, SCALING);
    gemm32_wmma_f16<<<2 * (512  / 16), 32, 0, stream>>>(hidden, k_w, k_b, kn, 512,  1024, 1.0f);
    gemm32_wmma_f16<<<2 * (1024 / 16), 32, 0, stream>>>(hidden, v_w, v_b, vn, 1024, 1024, 1.0f);

    // Fused flash-decode over the 805 MB KV stream (bandwidth-bound core).
    attn_decode<<<32 * 16, 256, 0, stream>>>(qs, kn, vn, past_key, past_value, mask, kvlen, attn);

    // Output projection straight into d_out.
    gemm32_wmma_f16<<<2 * (1024 / 16), 32, 0, stream>>>(attn, o_w, o_b, out, 1024, 1024, 1.0f);
}
